// MyModelV2_498216206514
// MI455X (gfx1250) — compile-verified, matrix operand order checked by reading, not run on hardware
//
#include <hip/hip_runtime.h>

// ---------------------------------------------------------------------------
// Stacked LSTM (units 256/128/64/32/20, F=128, B=512, T=1024) for gfx1250.
//  * pack_layer_kernel: [W;U] -> bf16 fragments in exact CDNA5 WMMA B-matrix
//    lane layout, gate-major so one wave owns all 4 gates of 16 units.
//  * lstm_stack_kernel: 1 block = 32 batch rows (2 WMMA M-tiles), 8 wave32.
//    Per k-tile: load 4 gate B-fragments once, apply to both M-tiles
//    (halves L2 weight traffic vs M=16 — the per-step bottleneck).
//    h (bf16) and c (f32) stay in LDS across the whole T loop.
// ---------------------------------------------------------------------------

typedef __attribute__((ext_vector_type(16))) __bf16 v16bf;
typedef __attribute__((ext_vector_type(8)))  __bf16 v8bf;
typedef __attribute__((ext_vector_type(8)))  float  v8f;

#define BLOCK   256
#define MTILE   32    // batch rows per block (2 WMMA M-tiles)
#define MSUB    2
#define HSTRIDE 512   // sum of padded units: 256+128+64+32+32
#define XSTRIDE 128

// Total packed weight elements (bf16) across 5 layers
#define WTOTAL  659456

__device__ __forceinline__ float fsig(float x) {
    return 1.0f / (1.0f + __expf(-x));
}
__device__ __forceinline__ float ftanh(float x) {
    float e = __expf(-2.0f * x);
    return (1.0f - e) / (1.0f + e);
}

// ---------------------------------------------------------------------------
// Weight pre-pack: fragment order F = (jt*4 + g)*nkt + kt, element order
// [lane][j] with lane = (half<<4)|col : N = jt*16+col, and in-lane j maps to
// K = kt*32 + half*8 + (j&7) + (j>=8 ? 16 : 0)   (16-bit B-matrix layout).
// ---------------------------------------------------------------------------
__global__ void pack_layer_kernel(const float* __restrict__ W,
                                  const float* __restrict__ U,
                                  const float* __restrict__ b,
                                  __bf16* __restrict__ wdst,
                                  float*  __restrict__ bdst,
                                  int Kin, int Ureal, int Upad,
                                  int nkt, int njt)
{
    int total = njt * 4 * nkt * 512;
    int stride = gridDim.x * blockDim.x;
    for (int idx = blockIdx.x * blockDim.x + threadIdx.x; idx < total; idx += stride) {
        int e    = idx & 511;
        int F    = idx >> 9;
        int kt   = F % nkt;
        int t2   = F / nkt;
        int g    = t2 & 3;
        int jt   = t2 >> 2;
        int lane = e >> 4;
        int j    = e & 15;
        int colt = lane & 15;
        int half = lane >> 4;
        int K    = kt * 32 + half * 8 + (j & 7) + ((j & 8) ? 16 : 0);
        int jc   = jt * 16 + colt;
        float v = 0.0f;
        if (jc < Ureal) {
            int n = g * Ureal + jc;           // column in the [*,4u] matrices
            if (K < Kin) {
                v = W[(size_t)K * (4 * Ureal) + n];
            } else {
                int kr = K - Kin;
                if (kr < Ureal) v = U[(size_t)kr * (4 * Ureal) + n];
            }
        }
        wdst[idx] = (__bf16)v;
    }
    // bias (padded to Upad, zero in padding)
    for (int idx = blockIdx.x * blockDim.x + threadIdx.x; idx < 4 * Upad; idx += stride) {
        int g  = idx / Upad;
        int jc = idx % Upad;
        bdst[idx] = (jc < Ureal) ? b[g * Ureal + jc] : 0.0f;
    }
}

// ---------------------------------------------------------------------------
// Main recurrence kernel.
// ---------------------------------------------------------------------------
__global__ __launch_bounds__(BLOCK) void lstm_stack_kernel(
    const float*  __restrict__ x,
    const __bf16* __restrict__ wfrag,
    const float*  __restrict__ bpack,
    float*        __restrict__ out,
    int T)
{
    const int KIN[5]     = {128, 256, 128, 64, 32};
    const int UPAD[5]    = {256, 128, 64, 32, 32};
    const int NKT[5]     = {12, 12, 6, 3, 2};       // (KIN+UPAD)/32
    const int NJT[5]     = {16, 8, 4, 2, 2};        // UPAD/16
    const int HOFF[5]    = {0, 256, 384, 448, 480};
    const size_t WOFF[5] = {0, 393216, 589824, 638976, 651264};
    const int BOFF[5]    = {0, 1024, 1536, 1792, 1920};

    __shared__ __bf16 xbuf[MTILE * XSTRIDE];   // 8 KB  : x_t tile (bf16)
    __shared__ __bf16 hbuf[MTILE * HSTRIDE];   // 32 KB : h state, all layers
    __shared__ float  cbuf[MTILE * HSTRIDE];   // 64 KB : c state, all layers

    const int tid  = threadIdx.x;
    const int wave = tid >> 5;
    const int lane = tid & 31;
    const int col  = lane & 15;
    const int half = lane >> 4;
    const int b0   = blockIdx.x * MTILE;

    for (int i = tid; i < MTILE * HSTRIDE; i += BLOCK) {
        hbuf[i] = (__bf16)0.0f;
        cbuf[i] = 0.0f;
    }
    __syncthreads();

    for (int t = 0; t < T; ++t) {
        // ---- stage x_t tile into LDS (f32 -> bf16), prefetch next step ----
        for (int i = tid; i < MTILE * XSTRIDE; i += BLOCK) {
            int m = i >> 7;
            int k = i & 127;
            const float* gp = x + ((size_t)(b0 + m) * T + t) * XSTRIDE + k;
            xbuf[i] = (__bf16)(*gp);
            if ((k & 15) == 0 && (t + 1) < T)
                __builtin_prefetch(gp + XSTRIDE, 0, 1);   // global_prefetch_b8
        }
        __syncthreads();

        #pragma unroll
        for (int l = 0; l < 5; ++l) {
            __bf16 hstage[2][MSUB][8];

            // ------- GEMM + gate phase (reads old hbuf[l], new hbuf[l-1]) ----
            #pragma unroll
            for (int pass = 0; pass < 2; ++pass) {
                int jt = wave + pass * 8;                 // wave-uniform guard
                if (jt < NJT[l]) {
                    v8f acc[MSUB][4] = {};
                    for (int kt = 0; kt < NKT[l]; ++kt) {
                        int kb = kt * 32;
                        const __bf16* abase;
                        int rs, kk;
                        if (kb < KIN[l]) {                // input part of A
                            if (l == 0) { abase = xbuf; rs = XSTRIDE; kk = kb; }
                            else        { abase = hbuf; rs = HSTRIDE; kk = HOFF[l - 1] + kb; }
                        } else {                          // recurrent part of A
                            abase = hbuf; rs = HSTRIDE; kk = HOFF[l] + (kb - KIN[l]);
                        }

                        // 4 gate B-fragments for this k-tile (reused by both
                        // M-tiles -> halves L2 weight traffic)
                        const __bf16* wp = wfrag + WOFF[l]
                            + ((size_t)((jt * 4 + 0) * NKT[l] + kt)) * 512
                            + (size_t)lane * 16;
                        v16bf bfr[4];
                        #pragma unroll
                        for (int g = 0; g < 4; ++g)
                            bfr[g] = *(const v16bf*)(wp + (size_t)g * NKT[l] * 512);

                        #pragma unroll
                        for (int mt = 0; mt < MSUB; ++mt) {
                            // A fragment: lane row M = mt*16+col -> 2x ds_load_b128
                            const __bf16* ap = abase
                                + (size_t)(mt * 16 + col) * rs + kk + half * 8;
                            union { v16bf v; v8bf h[2]; } afr;
                            afr.h[0] = *(const v8bf*)(ap);
                            afr.h[1] = *(const v8bf*)(ap + 16);
                            #pragma unroll
                            for (int g = 0; g < 4; ++g) {
                                acc[mt][g] = __builtin_amdgcn_wmma_f32_16x16x32_bf16(
                                    false, afr.v, false, bfr[g], (short)0,
                                    acc[mt][g], false, false);
                            }
                        }
                    }
                    // ---- gates: all four gates of unit j live in this lane ----
                    int j = jt * 16 + col;
                    float bi = bpack[BOFF[l] + 0 * UPAD[l] + j];
                    float bf = bpack[BOFF[l] + 1 * UPAD[l] + j];
                    float bg = bpack[BOFF[l] + 2 * UPAD[l] + j];
                    float bo = bpack[BOFF[l] + 3 * UPAD[l] + j];
                    #pragma unroll
                    for (int mt = 0; mt < MSUB; ++mt) {
                        #pragma unroll
                        for (int r = 0; r < 8; ++r) {
                            int m = mt * 16 + r + 8 * half;   // C-layout M
                            float iv = fsig (acc[mt][0][r] + bi);
                            float fv = fsig (acc[mt][1][r] + bf);
                            float gv = ftanh(acc[mt][2][r] + bg);
                            float ov = fsig (acc[mt][3][r] + bo);
                            float cc = cbuf[m * HSTRIDE + HOFF[l] + j];
                            cc = fv * cc + iv * gv;
                            cbuf[m * HSTRIDE + HOFF[l] + j] = cc;
                            hstage[pass][mt][r] = (__bf16)(ov * ftanh(cc));
                        }
                    }
                }
            }
            __syncthreads();   // all reads of old hbuf[l] complete

            // ------- publish new h for this layer ---------------------------
            #pragma unroll
            for (int pass = 0; pass < 2; ++pass) {
                int jt = wave + pass * 8;
                if (jt < NJT[l]) {
                    int j = jt * 16 + col;
                    #pragma unroll
                    for (int mt = 0; mt < MSUB; ++mt) {
                        #pragma unroll
                        for (int r = 0; r < 8; ++r) {
                            int m = mt * 16 + r + 8 * half;
                            hbuf[m * HSTRIDE + HOFF[l] + j] = hstage[pass][mt][r];
                        }
                    }
                }
            }
            __syncthreads();   // new h visible before next layer / next step
        }
    }

    // ---- write top-layer last hidden [MTILE, 20] as f32 ----
    for (int i = tid; i < MTILE * 20; i += BLOCK) {
        int m = i / 20;
        int j = i % 20;
        out[(size_t)(b0 + m) * 20 + j] = (float)hbuf[m * HSTRIDE + HOFF[4] + j];
    }
}

// ---------------------------------------------------------------------------
// Host-side launcher.
// d_in order: x, then (W,U,b) per layer. Output: [B,20] f32.
// Workspace: packed bf16 weights (WTOTAL elems) then packed f32 biases.
// ---------------------------------------------------------------------------
extern "C" void kernel_launch(void* const* d_in, const int* in_sizes, int n_in,
                              void* d_out, int out_size, void* d_ws, size_t ws_size,
                              hipStream_t stream)
{
    const float* x = (const float*)d_in[0];

    __bf16* wws = (__bf16*)d_ws;
    float*  bws = (float*)((char*)d_ws + (size_t)WTOTAL * 2);

    const int KIN[5]     = {128, 256, 128, 64, 32};
    const int UREAL[5]   = {256, 128, 64, 32, 20};
    const int UPAD[5]    = {256, 128, 64, 32, 32};
    const int NKT[5]     = {12, 12, 6, 3, 2};
    const int NJT[5]     = {16, 8, 4, 2, 2};
    const size_t WOFF[5] = {0, 393216, 589824, 638976, 651264};
    const int BOFF[5]    = {0, 1024, 1536, 1792, 1920};

    for (int l = 0; l < 5; ++l) {
        const float* W = (const float*)d_in[1 + 3 * l];
        const float* U = (const float*)d_in[2 + 3 * l];
        const float* b = (const float*)d_in[3 + 3 * l];
        int total  = NJT[l] * 4 * NKT[l] * 512;
        int blocks = (total + BLOCK - 1) / BLOCK;
        pack_layer_kernel<<<blocks, BLOCK, 0, stream>>>(
            W, U, b,
            wws + WOFF[l], bws + BOFF[l],
            KIN[l], UREAL[l], UPAD[l], NKT[l], NJT[l]);
    }

    int B = out_size / 20;                 // 512
    int T = in_sizes[0] / (B * XSTRIDE);   // 1024
    lstm_stack_kernel<<<B / MTILE, BLOCK, 0, stream>>>(
        x, wws, bws, (float*)d_out, T);
}
